// MoEPureField_10015863734692
// MI455X (gfx1250) — matrix operand order repaired and checked by reading, not compiled, not run on hardware
//
#include <hip/hip_runtime.h>

// ---------------------------------------------------------------------------
// MoE dense-expert FFN for MI455X (gfx1250, wave32, WMMA bf16 16x16x32).
//   gate:  s[n,e] = renorm(top5(softmax((x@gw+gb)/e))) * (+1 e<4, -1 else)
//   prep:  (if ws allows) x->bf16, w1->[e][h][k] bf16, w2->[e][o][k] bf16
//   ffn1:  H = relu(x @ w1[e] + b1[e])   (bf16, 2048-token chunk, L2-resident)
//   ffn2:  out (+)= s[:,e] * (H @ w2[e] + b2[e])
// Block tile 64x256, 8 waves (2x4), 2x4 WMMA tiles per wave, f32 accum.
// Double-buffered LDS; async global->LDS staging overlapped with WMMA via
// s_wait_asynccnt "keep-N" (async loads retire in order).
// ---------------------------------------------------------------------------

typedef __attribute__((ext_vector_type(16))) __bf16 v16bf;
typedef __attribute__((ext_vector_type(8)))  float  v8f;
typedef __attribute__((ext_vector_type(4)))  int    v4i;

#define D_IN   1024
#define D_HID  4096
#define D_OUT  1024
#define NEXP   8
#define NTOK   8192
#define NCH    2048
#define BM     64
#define BN     256
#define KC     64
#define LDK    72          // padded LDS row stride (bf16 elements)

union FragU { v16bf v; uint4 q[2]; };

__device__ __forceinline__ unsigned short f2bf(float f) {
  unsigned u = __builtin_bit_cast(unsigned, f);
  u += 0x7FFFu + ((u >> 16) & 1u);              // RNE
  return (unsigned short)(u >> 16);
}

#if __has_builtin(__builtin_amdgcn_cvt_pk_bf16_f32)
__device__ __forceinline__ unsigned f2bf2(float a, float b) {
  auto r = __builtin_amdgcn_cvt_pk_bf16_f32(a, b);   // v_cvt_pk_bf16_f32 (RNE)
  return __builtin_bit_cast(unsigned, r);
}
#else
__device__ __forceinline__ unsigned f2bf2(float a, float b) {
  return (unsigned)f2bf(a) | ((unsigned)f2bf(b) << 16);
}
#endif

#if __has_builtin(__builtin_amdgcn_global_load_async_to_lds_b128) && \
    __has_builtin(__builtin_amdgcn_s_wait_asynccnt)
#define HAVE_ASYNC 1
#endif

// 16-byte global->LDS tile copy (async DMA path when available)
__device__ __forceinline__ void copy16(unsigned short* ldst, const unsigned short* gsrc) {
#ifdef HAVE_ASYNC
  __builtin_amdgcn_global_load_async_to_lds_b128(
      (__attribute__((address_space(1))) v4i*)gsrc,
      (__attribute__((address_space(3))) v4i*)ldst, 0, 0);
#else
  *(uint4*)ldst = *(const uint4*)gsrc;
#endif
}
// wait until at most N async-to-LDS ops remain in flight (in-order retire)
template <int N>
__device__ __forceinline__ void wait_async_keep() {
#ifdef HAVE_ASYNC
  __builtin_amdgcn_s_wait_asynccnt(N);
#endif
}

// ---------------------------------------------------------------- gate ------
__global__ __launch_bounds__(256)
void moe_gate(const float* __restrict__ x, const float* __restrict__ gw,
              const float* __restrict__ gb, float* __restrict__ s) {
  const int wave = threadIdx.x >> 5, lane = threadIdx.x & 31;
  const int token = blockIdx.x * 8 + wave;
  const float* xr = x + (size_t)token * D_IN;

  float acc[NEXP];
#pragma unroll
  for (int e = 0; e < NEXP; ++e) acc[e] = 0.f;
  for (int d = lane; d < D_IN; d += 32) {
    float xv = xr[d];
    const float4* g = (const float4*)(gw + (size_t)d * NEXP);
    float4 g0 = g[0], g1 = g[1];
    acc[0] += xv * g0.x; acc[1] += xv * g0.y; acc[2] += xv * g0.z; acc[3] += xv * g0.w;
    acc[4] += xv * g1.x; acc[5] += xv * g1.y; acc[6] += xv * g1.z; acc[7] += xv * g1.w;
  }
#pragma unroll
  for (int off = 16; off > 0; off >>= 1)
#pragma unroll
    for (int e = 0; e < NEXP; ++e) acc[e] += __shfl_xor(acc[e], off, 32);

  if (lane == 0) {
    const float invT = 0.36787944117144233f;   // 1/e
    float sc[NEXP], m = -3.4e38f;
#pragma unroll
    for (int i = 0; i < NEXP; ++i) { sc[i] = (acc[i] + gb[i]) * invT; m = fmaxf(m, sc[i]); }
    float p[NEXP], den = 0.f;
#pragma unroll
    for (int i = 0; i < NEXP; ++i) { p[i] = expf(sc[i] - m); den += p[i]; }
    float inv = 1.f / den;
#pragma unroll
    for (int i = 0; i < NEXP; ++i) p[i] *= inv;
    float ksum = 0.f;
    bool keep[NEXP];
#pragma unroll
    for (int i = 0; i < NEXP; ++i) {
      int r = 0;
#pragma unroll
      for (int j = 0; j < NEXP; ++j)
        if (p[j] > p[i] || (p[j] == p[i] && j < i)) ++r;
      keep[i] = (r < 5);
      if (keep[i]) ksum += p[i];
    }
    float invk = 1.f / (ksum + 1e-8f);
    float* so = s + (size_t)token * NEXP;
#pragma unroll
    for (int i = 0; i < NEXP; ++i)
      so[i] = keep[i] ? (p[i] * invk * (i < 4 ? 1.f : -1.f)) : 0.f;
  }
}

// ------------------------------------------------- prep: fp32 -> bf16 -------
__global__ __launch_bounds__(256)
void cvt_x(const float* __restrict__ src, unsigned short* __restrict__ dst) {
  size_t i = ((size_t)blockIdx.x * 256 + threadIdx.x) * 4;
  float4 v = *(const float4*)(src + i);
  uint2 pk; pk.x = f2bf2(v.x, v.y); pk.y = f2bf2(v.z, v.w);
  *(uint2*)(dst + i) = pk;
}

// src [z][K][Nn] fp32 -> dst [z][Nn][K] bf16  (64x64 tiles through LDS)
__global__ __launch_bounds__(256)
void cvt_wT(const float* __restrict__ src, unsigned short* __restrict__ dst,
            int K, int Nn) {
  __shared__ unsigned short T[64 * LDK];
  const int tid = threadIdx.x;
  const int k0 = blockIdx.x * 64, n0 = blockIdx.y * 64;
  const size_t eoff = (size_t)blockIdx.z * K * Nn;

  const int c4 = tid & 15, r0 = tid >> 4;
#pragma unroll
  for (int i = 0; i < 4; ++i) {
    int r = r0 + 16 * i;
    float4 v = *(const float4*)(src + eoff + (size_t)(k0 + r) * Nn + n0 + 4 * c4);
    *(unsigned*)(&T[r * LDK + 4 * c4 + 0]) = f2bf2(v.x, v.y);
    *(unsigned*)(&T[r * LDK + 4 * c4 + 2]) = f2bf2(v.z, v.w);
  }
  __syncthreads();
  const int h = tid >> 2, qb = (tid & 3) * 2;
#pragma unroll
  for (int j = 0; j < 2; ++j) {
    int q = qb + j;
    union { uint4 u; unsigned short s[8]; } pk;
#pragma unroll
    for (int mm = 0; mm < 8; ++mm) pk.s[mm] = T[(8 * q + mm) * LDK + h];
    *(uint4*)(dst + eoff + (size_t)(n0 + h) * K + k0 + 8 * q) = pk.u;
  }
}

// ------------------------------------------------------- layer 1 ------------
template <bool PT>
__global__ __launch_bounds__(256)
void moe_ffn1(const float* __restrict__ x, const unsigned short* __restrict__ xbf,
              const float* __restrict__ w1, const unsigned short* __restrict__ w1t,
              const float* __restrict__ b1, unsigned short* __restrict__ Hb,
              int e, int tokBase) {
  __shared__ unsigned short Xs[2][BM * LDK];
  __shared__ unsigned short Ws[2][BN * LDK];

  const int tid = threadIdx.x;
  const int wave = tid >> 5, lane = tid & 31;
  const int wm = wave >> 2, wn = wave & 3;          // 2 x 4 waves
  const int ln = lane & 15, hl = lane >> 4;
  const int rowBase = blockIdx.x * BM;              // chunk-local tokens
  const int colBase = blockIdx.y * BN;              // hidden cols

  v8f acc[2][4] = {};

  const int q = tid & 7, r0 = tid >> 3;             // PT copy map
  const int xc4 = tid & 15, xr0 = tid >> 4;         // fallback A map
  const int wc4 = tid & 63, wr0 = tid >> 6;         // fallback B map

  auto stage = [&](int k0, int bi) {
    if constexpr (PT) {
#pragma unroll
      for (int i = 0; i < 2; ++i) {                 // A: bf16 x rows (async)
        int r = r0 + 32 * i;
        copy16(&Xs[bi][r * LDK + 8 * q],
               xbf + (size_t)(tokBase + rowBase + r) * D_IN + k0 + 8 * q);
      }
#pragma unroll
      for (int i = 0; i < 8; ++i) {                 // B: w1t[e][h][k] (async)
        int n = r0 + 32 * i;
        copy16(&Ws[bi][n * LDK + 8 * q],
               w1t + ((size_t)e * D_HID + colBase + n) * D_IN + k0 + 8 * q);
      }
    } else {
      const float* xblk = x + (size_t)(tokBase + rowBase) * D_IN;
#pragma unroll
      for (int i = 0; i < 4; ++i) {                 // A: fp32 -> bf16
        int r = xr0 + 16 * i;
        float4 v = *(const float4*)(xblk + (size_t)r * D_IN + k0 + 4 * xc4);
        uint2 pk; pk.x = f2bf2(v.x, v.y); pk.y = f2bf2(v.z, v.w);
        *(uint2*)(&Xs[bi][r * LDK + 4 * xc4]) = pk;
      }
      const float* w1e = w1 + (size_t)e * D_IN * D_HID;
#pragma unroll
      for (int i = 0; i < 16; ++i) {                // B: transpose during store
        int kr = wr0 + 4 * i;
        float4 v = *(const float4*)(w1e + (size_t)(k0 + kr) * D_HID + colBase + 4 * wc4);
        Ws[bi][(4 * wc4 + 0) * LDK + kr] = f2bf(v.x);
        Ws[bi][(4 * wc4 + 1) * LDK + kr] = f2bf(v.y);
        Ws[bi][(4 * wc4 + 2) * LDK + kr] = f2bf(v.z);
        Ws[bi][(4 * wc4 + 3) * LDK + kr] = f2bf(v.w);
      }
    }
  };

  auto compute = [&](int bi) {
#pragma unroll
    for (int kk = 0; kk < KC; kk += 32) {
      FragU a[2], b[4];
#pragma unroll
      for (int ti = 0; ti < 2; ++ti) {
        const unsigned short* p = &Xs[bi][(32 * wm + 16 * ti + ln) * LDK + kk + 8 * hl];
        a[ti].q[0] = *(const uint4*)p;
        a[ti].q[1] = *(const uint4*)(p + 16);
      }
#pragma unroll
      for (int tj = 0; tj < 4; ++tj) {
        const unsigned short* p = &Ws[bi][(64 * wn + 16 * tj + ln) * LDK + kk + 8 * hl];
        b[tj].q[0] = *(const uint4*)p;
        b[tj].q[1] = *(const uint4*)(p + 16);
      }
#pragma unroll
      for (int ti = 0; ti < 2; ++ti)
#pragma unroll
        for (int tj = 0; tj < 4; ++tj)
          acc[ti][tj] = __builtin_amdgcn_wmma_f32_16x16x32_bf16(
              false, a[ti].v, false, b[tj].v, (short)0, acc[ti][tj], false, false);
    }
  };

  constexpr int NST = PT ? 10 : 0;   // async ops per stage
  constexpr int NIT = D_IN / KC;
  stage(0, 0);
  for (int it = 0; it < NIT - 1; ++it) {
    int bi = it & 1;
    stage((it + 1) * KC, bi ^ 1);    // issue next chunk first ...
    wait_async_keep<NST>();          // ... then wait for current chunk only
    __syncthreads();
    compute(bi);
    __syncthreads();                 // all reads done before buffer is re-staged
  }
  wait_async_keep<0>();
  __syncthreads();
  compute((NIT - 1) & 1);

#pragma unroll
  for (int ti = 0; ti < 2; ++ti)
#pragma unroll
    for (int tj = 0; tj < 4; ++tj) {
      int h = colBase + 64 * wn + 16 * tj + ln;
      float bv = b1[(size_t)e * D_HID + h];
      int rl = rowBase + 32 * wm + 16 * ti + 8 * hl;
#pragma unroll
      for (int v = 0; v < 8; ++v) {
        float val = acc[ti][tj][v] + bv;
        val = val > 0.f ? val : 0.f;
        Hb[(size_t)(rl + v) * D_HID + h] = f2bf(val);
      }
    }
}

// ------------------------------------------------------- layer 2 ------------
template <bool PT>
__global__ __launch_bounds__(256)
void moe_ffn2(const unsigned short* __restrict__ Hb,
              const float* __restrict__ w2, const unsigned short* __restrict__ w2t,
              const float* __restrict__ b2, const float* __restrict__ s,
              float* __restrict__ out, int e, int tokBase, int first) {
  __shared__ unsigned short Xs[2][BM * LDK];
  __shared__ unsigned short Ws[2][BN * LDK];

  const int tid = threadIdx.x;
  const int wave = tid >> 5, lane = tid & 31;
  const int wm = wave >> 2, wn = wave & 3;
  const int ln = lane & 15, hl = lane >> 4;
  const int rowBase = blockIdx.x * BM;
  const int colBase = blockIdx.y * BN;

  v8f acc[2][4] = {};

  const int q = tid & 7, r0 = tid >> 3;
  const int wc4 = tid & 63, wr0 = tid >> 6;

  auto stage = [&](int k0, int bi) {
#pragma unroll
    for (int i = 0; i < 2; ++i) {                   // A: bf16 H rows (async)
      int r = r0 + 32 * i;
      copy16(&Xs[bi][r * LDK + 8 * q],
             Hb + (size_t)(rowBase + r) * D_HID + k0 + 8 * q);
    }
    if constexpr (PT) {
#pragma unroll
      for (int i = 0; i < 8; ++i) {                 // B: w2t[e][o][k] (async)
        int n = r0 + 32 * i;
        copy16(&Ws[bi][n * LDK + 8 * q],
               w2t + ((size_t)e * D_OUT + colBase + n) * D_HID + k0 + 8 * q);
      }
    } else {
      const float* w2e = w2 + (size_t)e * D_HID * D_OUT;
#pragma unroll
      for (int i = 0; i < 16; ++i) {                // B: transpose during store
        int kr = wr0 + 4 * i;
        float4 v = *(const float4*)(w2e + (size_t)(k0 + kr) * D_OUT + colBase + 4 * wc4);
        Ws[bi][(4 * wc4 + 0) * LDK + kr] = f2bf(v.x);
        Ws[bi][(4 * wc4 + 1) * LDK + kr] = f2bf(v.y);
        Ws[bi][(4 * wc4 + 2) * LDK + kr] = f2bf(v.z);
        Ws[bi][(4 * wc4 + 3) * LDK + kr] = f2bf(v.w);
      }
    }
  };

  auto compute = [&](int bi) {
#pragma unroll
    for (int kk = 0; kk < KC; kk += 32) {
      FragU a[2], b[4];
#pragma unroll
      for (int ti = 0; ti < 2; ++ti) {
        const unsigned short* p = &Xs[bi][(32 * wm + 16 * ti + ln) * LDK + kk + 8 * hl];
        a[ti].q[0] = *(const uint4*)p;
        a[ti].q[1] = *(const uint4*)(p + 16);
      }
#pragma unroll
      for (int tj = 0; tj < 4; ++tj) {
        const unsigned short* p = &Ws[bi][(64 * wn + 16 * tj + ln) * LDK + kk + 8 * hl];
        b[tj].q[0] = *(const uint4*)p;
        b[tj].q[1] = *(const uint4*)(p + 16);
      }
#pragma unroll
      for (int ti = 0; ti < 2; ++ti)
#pragma unroll
        for (int tj = 0; tj < 4; ++tj)
          acc[ti][tj] = __builtin_amdgcn_wmma_f32_16x16x32_bf16(
              false, a[ti].v, false, b[tj].v, (short)0, acc[ti][tj], false, false);
    }
  };

  constexpr int NST = PT ? 10 : 2;   // async ops per stage
  constexpr int NIT = D_HID / KC;
  stage(0, 0);
  for (int it = 0; it < NIT - 1; ++it) {
    int bi = it & 1;
    stage((it + 1) * KC, bi ^ 1);
    wait_async_keep<NST>();
    __syncthreads();
    compute(bi);
    __syncthreads();
  }
  wait_async_keep<0>();
  __syncthreads();
  compute((NIT - 1) & 1);

#pragma unroll
  for (int ti = 0; ti < 2; ++ti)
#pragma unroll
    for (int v = 0; v < 8; ++v) {
      int tok = tokBase + rowBase + 32 * wm + 16 * ti + 8 * hl + v;
      float sv = s[(size_t)tok * NEXP + e];
#pragma unroll
      for (int tj = 0; tj < 4; ++tj) {
        int o = colBase + 64 * wn + 16 * tj + ln;
        float val = sv * (acc[ti][tj][v] + b2[(size_t)e * D_OUT + o]);
        float* po = out + (size_t)tok * D_OUT + o;
        if (first) *po = val;
        else       *po += val;
      }
    }
}

// ---------------------------------------------------------------------------
extern "C" void kernel_launch(void* const* d_in, const int* in_sizes, int n_in,
                              void* d_out, int out_size, void* d_ws, size_t ws_size,
                              hipStream_t stream) {
  (void)in_sizes; (void)n_in; (void)out_size;
  const float* x  = (const float*)d_in[0];
  const float* gw = (const float*)d_in[1];
  const float* gb = (const float*)d_in[2];
  const float* w1 = (const float*)d_in[3];
  const float* b1 = (const float*)d_in[4];
  const float* w2 = (const float*)d_in[5];
  const float* b2 = (const float*)d_in[6];
  float* out = (float*)d_out;

  char* wsp = (char*)d_ws;
  size_t off = 0;
  float* s = (float*)(wsp + off);            off += (size_t)NTOK * NEXP * 4;
  unsigned short* Hb  = (unsigned short*)(wsp + off); off += (size_t)NCH * D_HID * 2;
  unsigned short* xbf = (unsigned short*)(wsp + off); off += (size_t)NTOK * D_IN * 2;
  unsigned short* w1t = (unsigned short*)(wsp + off); off += (size_t)NEXP * D_IN * D_HID * 2;
  unsigned short* w2t = (unsigned short*)(wsp + off); off += (size_t)NEXP * D_HID * D_OUT * 2;
  const bool pt = (ws_size >= off);          // enough scratch for bf16 mirrors?

  if (pt) {
    cvt_x<<<NTOK * D_IN / (256 * 4), 256, 0, stream>>>(x, xbf);
    cvt_wT<<<dim3(D_IN / 64, D_HID / 64, NEXP), 256, 0, stream>>>(w1, w1t, D_IN, D_HID);
    cvt_wT<<<dim3(D_HID / 64, D_OUT / 64, NEXP), 256, 0, stream>>>(w2, w2t, D_HID, D_OUT);
  }
  moe_gate<<<NTOK / 8, 256, 0, stream>>>(x, gw, gb, s);

  for (int c = 0; c < NTOK / NCH; ++c) {
    int tokBase = c * NCH;
    for (int e = 0; e < NEXP; ++e) {
      if (pt) {
        moe_ffn1<true><<<dim3(NCH / BM, D_HID / BN), 256, 0, stream>>>(
            x, xbf, w1, w1t, b1, Hb, e, tokBase);
        moe_ffn2<true><<<dim3(NCH / BM, D_OUT / BN), 256, 0, stream>>>(
            Hb, w2, w2t, b2, s, out, e, tokBase, e == 0);
      } else {
        moe_ffn1<false><<<dim3(NCH / BM, D_HID / BN), 256, 0, stream>>>(
            x, xbf, w1, w1t, b1, Hb, e, tokBase);
        moe_ffn2<false><<<dim3(NCH / BM, D_OUT / BN), 256, 0, stream>>>(
            Hb, w2, w2t, b2, s, out, e, tokBase, e == 0);
      }
    }
  }
}
